// NVP_29987461661108
// MI455X (gfx1250) — matrix operand order repaired and checked by reading, not compiled, not run on hardware
//
#include <hip/hip_runtime.h>
#include <hip/hip_bf16.h>

// ---------------------------------------------------------------------------
// RealNVP-style flow, fully fused for MI455X (gfx1250, wave32, WMMA bf16).
//
// Input flattening assumption (JAX pytree order, dict keys sorted):
//   d_in[0] = F (1,8,256)        d_in[1] = x (1,4096,8,3)
//   then params: for layer l in 0..2 (stride 28, base 2 + 28*l):
//     code: +0 W0(256x256) +1 b0 +2 W1(256x256) +3 b1
//     l1 (base +4), l2 (base +16), each 12 arrays:
//       proj: Wp1(3x512) bp1 Wp2(512x256) bp2
//       s:    Ws1(512x256) bs1 Ws2(256x3) bs2
//       t:    Wt1(512x256) bt1 Wt2(256x3) bt2
//   scales at 86: sW0(256x256) 87: sb0 88: sW1(256x3) 89: sb1
//
// d_ws usage (~3.4 MB): Fi[3][8][256] f32, inv_sigma[8][3] f32,
//   cbias[12][8][256] f32 (Fi @ Ws1/Wt1[0:256] + b1, per layer/cpl/{s,t}),
//   plus all big weights converted to bf16 in WMMA B-fragment order.
//
// NOTE: the reference's coupling masks are np.random.default_rng(0)
// .permutation(3) (a host constant, not an input). SEL below encodes the
// "kept" coordinate per flow layer; adjust if the permutation differs.
// ---------------------------------------------------------------------------

typedef __attribute__((ext_vector_type(16))) __bf16 v16bf;
typedef __attribute__((ext_vector_type(8)))  float  v8f;

__device__ __forceinline__ unsigned short f2bf(float f) {
  union { float f; unsigned int u; } v; v.f = f;
  unsigned int r = v.u + 0x7FFFu + ((v.u >> 16) & 1u);  // RNE
  return (unsigned short)(r >> 16);
}

// Position of element (row m, k) in the 16-bit A-matrix fragment stream:
// storage index = ((k/32)*32 + lane)*16 + j  with lane = m + 16*g,
// g = bit3 of (k%32), j = (k%32 & 7) + 8*bit4  (matches ISA 16x32 A layout).
__device__ __forceinline__ int afrag_pos(int m, int k) {
  int kt = k >> 5, kk = k & 31;
  int gg = (kk >> 3) & 1;
  int j  = (kk & 7) + ((kk & 16) >> 1);
  return ((kt * 32) + (m + (gg << 4))) * 16 + j;
}

// ------------------------------- prep kernels ------------------------------

// Fi = (leaky_relu(F@W0+b0, .01)) @ W1 + b1  -> (8,256)
__global__ __launch_bounds__(256) void fi_kernel(const float* F, const float* W0,
                                                 const float* b0, const float* W1,
                                                 const float* b1, float* out) {
  __shared__ float h[8 * 256];
  const int t = threadIdx.x;
  for (int mm = 0; mm < 8; ++mm) {
    float acc = b0[t];
    for (int k = 0; k < 256; ++k) acc += F[mm * 256 + k] * W0[k * 256 + t];
    h[mm * 256 + t] = acc > 0.f ? acc : 0.01f * acc;
  }
  __syncthreads();
  for (int mm = 0; mm < 8; ++mm) {
    float acc = b1[t];
    for (int k = 0; k < 256; ++k) acc += h[mm * 256 + k] * W1[k * 256 + t];
    out[mm * 256 + t] = acc;
  }
}

// inv_sigma = 1 / (elu(relu(F@sW0+sb0)@sW1+sb1) + 1)   -> (8,3)
__global__ __launch_bounds__(256) void sigma_kernel(const float* F, const float* W0,
                                                    const float* b0, const float* W1,
                                                    const float* b1, float* invsig) {
  __shared__ float h[8 * 256];
  const int t = threadIdx.x;
  for (int mm = 0; mm < 8; ++mm) {
    float acc = b0[t];
    for (int k = 0; k < 256; ++k) acc += F[mm * 256 + k] * W0[k * 256 + t];
    h[mm * 256 + t] = acc > 0.f ? acc : 0.f;
  }
  __syncthreads();
  if (t < 24) {
    const int mm = t / 3, o = t % 3;
    float acc = b1[o];
    for (int k = 0; k < 256; ++k) acc += h[mm * 256 + k] * W1[k * 3 + o];
    const float sig = acc > 0.f ? (acc + 1.f) : __expf(acc);  // elu(z)+1
    invsig[t] = 1.f / sig;
  }
}

// cbias = Fi @ W1[0:256,:] + b1   (Fi half of the s/t first layer) -> (8,256)
__global__ __launch_bounds__(256) void cbias_kernel(const float* Fi, const float* W1,
                                                    const float* b1, float* out) {
  const int t = threadIdx.x;
  for (int mm = 0; mm < 8; ++mm) {
    float acc = b1[t];
    for (int k = 0; k < 256; ++k) acc += Fi[mm * 256 + k] * W1[k * 256 + t];
    out[mm * 256 + t] = acc;
  }
}

// Convert a row-major (K x nCols) f32 weight into bf16 B-fragment stream:
// dst[((nt*KT+kt)*32 + lane)*16 + j] = W[k, n],  k = kt*32 + 8*(lane/16) + (j&7) + 16*(j/8),
// n = nt*16 + lane%16 (zero-pad n >= nCols, used for the 256x3 heads).
__global__ void frag_kernel(const float* src, int srcLd, int nCols, int KT,
                            int total, unsigned short* dst) {
  const int e = blockIdx.x * blockDim.x + threadIdx.x;
  if (e >= total) return;
  const int j = e & 15, L = (e >> 4) & 31, tt = e >> 9;
  const int kt = tt % KT, nt = tt / KT;
  const int k = kt * 32 + ((L >> 4) << 3) + (j & 7) + ((j >> 3) << 4);
  const int n = nt * 16 + (L & 15);
  const float v = (n < nCols) ? src[k * srcLd + n] : 0.f;
  dst[e] = f2bf(v);
}

// ------------------------------- main kernel -------------------------------

struct MainArgs {
  const float* x;
  float* out;
  const float* Wp1[6];
  const float* bp1[6];
  const float* bp2[6];
  const float* bs2[6];
  const float* bt2[6];
  const float* cbias;      // [12][8][256] index (ci*2 + st)
  const float* inv_sigma;  // [8][3]
  const unsigned short* wp2frag;  // [6]  512x256 bf16 frags
  const unsigned short* w1low;    // [12] 256x256 bf16 frags (rows 256:512 of W1)
  const unsigned short* w2pad;    // [12] 256x16  bf16 frags (cols 3:16 zero)
};

__global__ __launch_bounds__(512) void nvp_main(MainArgs A) {
  __shared__ __align__(32) unsigned short A_h1[16 * 512];   // 16KB, A-frag layout
  __shared__ __align__(32) unsigned short A_proj[16 * 256]; // 8KB
  __shared__ __align__(32) unsigned short A_hs[16 * 256];   // 8KB
  __shared__ __align__(32) unsigned short A_ht[16 * 256];   // 8KB
  __shared__ float yt[48];
  __shared__ float s_arr[48];
  __shared__ float t_arr[48];

  const int tid = threadIdx.x;
  const int L = tid & 31;        // lane (wave32)
  const int wave = tid >> 5;     // 16 waves
  const int g = L >> 4;
  const int ln = L & 15;
  const int p0 = blockIdx.x * 16;  // first point of this 16-row tile

  if (tid < 48) yt[tid] = A.x[p0 * 3 + tid];
  __syncthreads();

  // SEL[l] = coordinate kept by MASKS2 in flow layer l
  // (must equal np.random.default_rng(0).permutation(3) — see header note)
  const int SEL[3] = {2, 0, 1};

  for (int layer = 0; layer < 3; ++layer) {
    const int c = SEL[layer];
    for (int cpl = 0; cpl < 2; ++cpl) {
      const int ci = layer * 2 + cpl;

      // ---- Phase A: h1 = relu(y1 @ Wp1 + bp1), K=3, one column per thread
      {
        const float* Wp1 = A.Wp1[ci];
        const float w0 = Wp1[tid], w1 = Wp1[512 + tid], w2 = Wp1[1024 + tid];
        const float bb = A.bp1[ci][tid];
        const bool k0 = (cpl == 0) ? (0 != c) : (0 == c);
        const bool k1 = (cpl == 0) ? (1 != c) : (1 == c);
        const bool k2 = (cpl == 0) ? (2 != c) : (2 == c);
#pragma unroll
        for (int m = 0; m < 16; ++m) {
          const float y0 = k0 ? yt[m * 3 + 0] : 0.f;
          const float y1 = k1 ? yt[m * 3 + 1] : 0.f;
          const float y2 = k2 ? yt[m * 3 + 2] : 0.f;
          float h = bb + y0 * w0 + y1 * w1 + y2 * w2;
          h = h > 0.f ? h : 0.f;
          A_h1[afrag_pos(m, tid)] = f2bf(h);
        }
      }
      __syncthreads();

      // ---- Phase B: proj = h1 @ Wp2 + bp2  (16x512 @ 512x256, wave = n-tile)
      {
        const unsigned short* wB = A.wp2frag + ci * (512 * 256);
        v8f acc = {};
        for (int kt = 0; kt < 16; ++kt) {
          v16bf a  = *(const v16bf*)&A_h1[(kt * 32 + L) * 16];
          v16bf bm = *(const v16bf*)&wB[((wave * 16 + kt) * 32 + L) * 16];
          acc = __builtin_amdgcn_wmma_f32_16x16x32_bf16(false, a, false, bm,
                                                        (short)0, acc, false, false);
        }
        const float bias = A.bp2[ci][wave * 16 + ln];
#pragma unroll
        for (int r = 0; r < 8; ++r)
          A_proj[afrag_pos(r + 8 * g, wave * 16 + ln)] = f2bf(acc[r] + bias);
      }
      __syncthreads();

      // ---- Phase C: s/t hidden = leaky_relu(proj @ W1low + cbias[m])
      for (int task = wave; task < 32; task += 16) {
        const int st = task >> 4;   // 0 = s, 1 = t
        const int nt = task & 15;
        const unsigned short* wB = A.w1low + (ci * 2 + st) * (256 * 256);
        v8f acc = {};
        for (int kt = 0; kt < 8; ++kt) {
          v16bf a  = *(const v16bf*)&A_proj[(kt * 32 + L) * 16];
          v16bf bm = *(const v16bf*)&wB[((nt * 8 + kt) * 32 + L) * 16];
          acc = __builtin_amdgcn_wmma_f32_16x16x32_bf16(false, a, false, bm,
                                                        (short)0, acc, false, false);
        }
        const float* cb = A.cbias + (ci * 2 + st) * 2048;
        const int n = nt * 16 + ln;
        unsigned short* dst = st ? A_ht : A_hs;
#pragma unroll
        for (int r = 0; r < 8; ++r) {
          // row m = r + 8g, its batch-m index = m % 8 = r
          float v = acc[r] + cb[r * 256 + n];
          v = v > 0.f ? v : 0.01f * v;
          dst[afrag_pos(r + 8 * g, n)] = f2bf(v);
        }
      }
      __syncthreads();

      // ---- Phase D: heads (256 -> 3, N padded to 16). Waves 0 (s) and 1 (t).
      if (wave < 2) {
        const int st = wave;
        const unsigned short* wB = A.w2pad + (ci * 2 + st) * 4096;
        const unsigned short* Ah = st ? A_ht : A_hs;
        v8f acc = {};
        for (int kt = 0; kt < 8; ++kt) {
          v16bf a  = *(const v16bf*)&Ah[(kt * 32 + L) * 16];
          v16bf bm = *(const v16bf*)&wB[(kt * 32 + L) * 16];
          acc = __builtin_amdgcn_wmma_f32_16x16x32_bf16(false, a, false, bm,
                                                        (short)0, acc, false, false);
        }
        if (ln < 3) {
          const float bb = (st ? A.bt2[ci] : A.bs2[ci])[ln];
          float* outv = st ? t_arr : s_arr;
#pragma unroll
          for (int r = 0; r < 8; ++r) {
            float v = acc[r] + bb;
            if (!st) v = fminf(fmaxf(v, -10.f), 10.f);  // clip s to [HT_MIN,HT_MAX]
            outv[(r + 8 * g) * 3 + ln] = v;
          }
        }
      }
      __syncthreads();

      // ---- Phase E: y = y1 + (1-mask) * ((y - t) * exp(-s))
      if (tid < 48) {
        const int i = tid % 3;
        const bool upd = (cpl == 0) ? (i == c) : (i != c);
        if (upd) yt[tid] = (yt[tid] - t_arr[tid]) * __expf(-s_arr[tid]);
      }
      __syncthreads();
    }
  }

  if (tid < 48) {
    const int row = tid / 3, i = tid % 3;
    A.out[p0 * 3 + tid] = yt[tid] * A.inv_sigma[(row & 7) * 3 + i];
  }
}

// ------------------------------- launcher ----------------------------------

extern "C" void kernel_launch(void* const* d_in, const int* in_sizes, int n_in,
                              void* d_out, int out_size, void* d_ws, size_t ws_size,
                              hipStream_t stream) {
  (void)in_sizes; (void)n_in; (void)out_size; (void)ws_size;

  const float* F = (const float*)d_in[0];
  const float* x = (const float*)d_in[1];
  auto P = [&](int i) { return (const float*)d_in[i]; };

  // workspace layout (needs ~3.4 MB)
  float* Fi = (float*)d_ws;                        // 3*2048 f32
  float* invsig = Fi + 3 * 2048;                   // 24 f32 (padded to 32)
  float* cbias = invsig + 32;                      // 12*2048 f32
  unsigned short* wp2frag = (unsigned short*)(cbias + 12 * 2048);  // 6*131072
  unsigned short* w1low = wp2frag + 6 * 512 * 256;                 // 12*65536
  unsigned short* w2pad = w1low + 12 * 256 * 256;                  // 12*4096

  // Fi per flow layer, then sigma
  for (int l = 0; l < 3; ++l) {
    const int cb = 2 + 28 * l;  // code block
    fi_kernel<<<1, 256, 0, stream>>>(F, P(cb + 0), P(cb + 1), P(cb + 2), P(cb + 3),
                                     Fi + l * 2048);
  }
  sigma_kernel<<<1, 256, 0, stream>>>(F, P(86), P(87), P(88), P(89), invsig);

  MainArgs MA;
  MA.x = x; MA.out = (float*)d_out;
  MA.cbias = cbias; MA.inv_sigma = invsig;
  MA.wp2frag = wp2frag; MA.w1low = w1low; MA.w2pad = w2pad;

  for (int l = 0; l < 3; ++l) {
    for (int cpl = 0; cpl < 2; ++cpl) {
      const int ci = l * 2 + cpl;
      const int cb = 2 + 28 * l + 4 + 12 * cpl;  // coupling base index
      MA.Wp1[ci] = P(cb + 0); MA.bp1[ci] = P(cb + 1); MA.bp2[ci] = P(cb + 3);
      MA.bs2[ci] = P(cb + 7); MA.bt2[ci] = P(cb + 11);

      // Fi-derived biases for the s/t first layers (rows 0:256 of Ws1/Wt1)
      cbias_kernel<<<1, 256, 0, stream>>>(Fi + l * 2048, P(cb + 4), P(cb + 5),
                                          cbias + (ci * 2 + 0) * 2048);
      cbias_kernel<<<1, 256, 0, stream>>>(Fi + l * 2048, P(cb + 8), P(cb + 9),
                                          cbias + (ci * 2 + 1) * 2048);

      // bf16 B-fragment conversions (L2-resident, reused by all 2048 blocks)
      frag_kernel<<<512, 256, 0, stream>>>(P(cb + 2), 256, 256, 16, 512 * 256,
                                           wp2frag + ci * 512 * 256);            // Wp2
      frag_kernel<<<256, 256, 0, stream>>>(P(cb + 4) + 256 * 256, 256, 256, 8,
                                           256 * 256, w1low + (ci * 2 + 0) * 65536); // Ws1 low
      frag_kernel<<<256, 256, 0, stream>>>(P(cb + 8) + 256 * 256, 256, 256, 8,
                                           256 * 256, w1low + (ci * 2 + 1) * 65536); // Wt1 low
      frag_kernel<<<16, 256, 0, stream>>>(P(cb + 6), 3, 3, 8, 4096,
                                          w2pad + (ci * 2 + 0) * 4096);           // Ws2 pad
      frag_kernel<<<16, 256, 0, stream>>>(P(cb + 10), 3, 3, 8, 4096,
                                          w2pad + (ci * 2 + 1) * 4096);           // Wt2 pad
    }
  }

  nvp_main<<<2048, 512, 0, stream>>>(MA);
}
